// GNNStack_12747462934829
// MI455X (gfx1250) — matrix operand order repaired and checked by reading, not compiled, not run on hardware
//
#include <hip/hip_runtime.h>
#include <math.h>

// ---------------------------------------------------------------------------
// Types for CDNA5 WMMA (wave32, 16x16x32 bf16 -> f32)
// ---------------------------------------------------------------------------
typedef __bf16 bf16t;
typedef bf16t v16bf __attribute__((ext_vector_type(16)));
typedef float v8f   __attribute__((ext_vector_type(8)));

union FragAB { v16bf v; unsigned int u[8]; };
union FragC  { v8f   v; float        f[8]; };

__device__ __forceinline__ unsigned short f2bf(float f) {
    unsigned int u = __float_as_uint(f);
    u += 0x7FFFu + ((u >> 16) & 1u);          // round to nearest even
    return (unsigned short)(u >> 16);
}

// Monotonic float <-> u32 encoding so unsigned atomicMax == float max
__device__ __forceinline__ unsigned int encf(float f) {
    unsigned int u = __float_as_uint(f);
    return (u & 0x80000000u) ? ~u : (u | 0x80000000u);
}
__device__ __forceinline__ float decf(unsigned int e) {
    unsigned int u = (e & 0x80000000u) ? (e & 0x7FFFFFFFu) : ~e;
    return __uint_as_float(u);
}
#define ENC_NEG_INF 0x007FFFFFu   /* encf(-inf) */

__device__ __forceinline__ float lrelu(float v) { return v > 0.f ? v : 0.2f * v; }

// ---------------------------------------------------------------------------
// GEMM: out[r, n] = sum_k in[r,k] * w[n,k] + bias[n]   (w row-major [NC,K])
// Optional relu on input load; optional fused attention coefficient products
// alh = al[n]*out, arh = ar[n]*out (GAT layers only).
// One wave per 16x16 output tile; block = (NCpad/16) waves covering 16 rows.
// ---------------------------------------------------------------------------
__global__ void gemm_wmma_kernel(const float* __restrict__ in,
                                 const float* __restrict__ w,
                                 const float* __restrict__ bias,
                                 const float* __restrict__ al,
                                 const float* __restrict__ ar,
                                 float* __restrict__ out,
                                 float* __restrict__ alh,
                                 float* __restrict__ arh,
                                 int nrows, int K, int NC, int NCpad,
                                 int ldo, int relu_in)
{
    __shared__ __align__(16) unsigned short sA[16 * 128];   // 4 KB
    __shared__ __align__(16) unsigned short sB[128 * 128];  // 32 KB

    const int rowBase = blockIdx.x * 16;

    // Stage A tile (16 x K) as bf16, zero-padded past nrows, relu-on-load.
    for (int idx = threadIdx.x; idx < 16 * K; idx += blockDim.x) {
        int r = idx / K, k = idx - r * K;
        int gr = rowBase + r;
        float v = 0.f;
        if (gr < nrows) {
            v = in[(size_t)gr * K + k];
            if (relu_in && v < 0.f) v = 0.f;
        }
        sA[idx] = f2bf(v);
    }
    // Stage B tile (NCpad x K) as bf16 (B[k][n] = w[n][k]), zero-pad n >= NC.
    for (int idx = threadIdx.x; idx < NCpad * K; idx += blockDim.x) {
        int n = idx / K, k = idx - n * K;
        float v = (n < NC) ? w[(size_t)n * K + k] : 0.f;
        sB[idx] = f2bf(v);
    }
    __syncthreads();

    const int lane    = threadIdx.x & 31;
    const int wave    = threadIdx.x >> 5;
    const int colBase = wave * 16;
    const int hi      = lane >> 4;     // K-half selector
    const int rc      = lane & 15;     // A-row / B-col inside tile

    v8f acc = {};
    for (int k0 = 0; k0 < K; k0 += 32) {
        FragAB a, b;
        // A layout: lane rc holds row rc; elem e -> K = (e&7) + 8*hi + 16*(e>=8)
        const unsigned short* arow = &sA[rc * K + k0 + hi * 8];
        // B layout: lane rc holds col rc; elem e -> K = (e&7) + 8*hi + 16*(e>=8)
        const unsigned short* brow = &sB[(colBase + rc) * K + k0 + hi * 8];
#pragma unroll
        for (int t = 0; t < 8; ++t) {
            int ka = 2 * t + ((t >= 4) ? 8 : 0);               // even, 4B aligned
            a.u[t] = *(const unsigned int*)(arow + ka);
            b.u[t] = *(const unsigned int*)(brow + ka);
        }
        acc = __builtin_amdgcn_wmma_f32_16x16x32_bf16(
                  false, a.v, false, b.v, (short)0, acc, false, false);
    }

    // C/D layout: row = v + 8*hi, col = rc
    FragC cu; cu.v = acc;
    const int ocol = colBase + rc;
#pragma unroll
    for (int v = 0; v < 8; ++v) {
        int orow = rowBase + v + hi * 8;
        if (orow < nrows && ocol < NC) {
            float val = cu.f[v] + bias[ocol];
            out[(size_t)orow * ldo + ocol] = val;
            if (alh) {
                alh[(size_t)orow * ldo + ocol] = al[ocol] * val;
                arh[(size_t)orow * ldo + ocol] = ar[ocol] * val;
            }
        }
    }
}

// ---------------------------------------------------------------------------
// Per-layer scratch init: m = -inf (encoded), s = 0, agg = 0
// ---------------------------------------------------------------------------
__global__ void init_kernel(unsigned int* __restrict__ mu,
                            float* __restrict__ ssum,
                            float* __restrict__ agg,
                            unsigned long long n)
{
    unsigned long long i = (unsigned long long)blockIdx.x * blockDim.x + threadIdx.x;
    if (i < n) { mu[i] = ENC_NEG_INF; ssum[i] = 0.f; agg[i] = 0.f; }
}

// ---------------------------------------------------------------------------
// Edge pass 1: segment max of leaky_relu(alh[src] + arh[dst]) per (dst, ch)
// One wave per edge, float4 per lane (128 channels).
// ---------------------------------------------------------------------------
__global__ void edge_max_kernel(const int* __restrict__ src, const int* __restrict__ dst,
                                const float* __restrict__ alh, const float* __restrict__ arh,
                                unsigned int* __restrict__ mu, int E)
{
    int e = blockIdx.x * (blockDim.x >> 5) + (threadIdx.x >> 5);
    if (e >= E) return;
    int lane = threadIdx.x & 31;
    int s = src[e], d = dst[e];
    const float4 a = *(const float4*)(alh + (size_t)s * 128 + lane * 4);
    const float4 r = *(const float4*)(arh + (size_t)d * 128 + lane * 4);
    unsigned int* mp = mu + (size_t)d * 128 + lane * 4;
    atomicMax(mp + 0, encf(lrelu(a.x + r.x)));
    atomicMax(mp + 1, encf(lrelu(a.y + r.y)));
    atomicMax(mp + 2, encf(lrelu(a.z + r.z)));
    atomicMax(mp + 3, encf(lrelu(a.w + r.w)));
}

// ---------------------------------------------------------------------------
// Edge pass 2: s[dst,ch] += exp(e - m[dst,ch])
// ---------------------------------------------------------------------------
__global__ void edge_expsum_kernel(const int* __restrict__ src, const int* __restrict__ dst,
                                   const float* __restrict__ alh, const float* __restrict__ arh,
                                   const unsigned int* __restrict__ mu,
                                   float* __restrict__ ssum, int E)
{
    int e = blockIdx.x * (blockDim.x >> 5) + (threadIdx.x >> 5);
    if (e >= E) return;
    int lane = threadIdx.x & 31;
    int s = src[e], d = dst[e];
    const float4 a = *(const float4*)(alh + (size_t)s * 128 + lane * 4);
    const float4 r = *(const float4*)(arh + (size_t)d * 128 + lane * 4);
    const uint4  m = *(const uint4*)(mu + (size_t)d * 128 + lane * 4);
    float* sp = ssum + (size_t)d * 128 + lane * 4;
    atomicAdd(sp + 0, __expf(lrelu(a.x + r.x) - decf(m.x)));
    atomicAdd(sp + 1, __expf(lrelu(a.y + r.y) - decf(m.y)));
    atomicAdd(sp + 2, __expf(lrelu(a.z + r.z) - decf(m.z)));
    atomicAdd(sp + 3, __expf(lrelu(a.w + r.w) - decf(m.w)));
}

// ---------------------------------------------------------------------------
// Edge pass 3: agg[dst,ch] += (exp(e - m) / (s + 1e-16)) * h[src,ch]
// ---------------------------------------------------------------------------
__global__ void edge_aggregate_kernel(const int* __restrict__ src, const int* __restrict__ dst,
                                      const float* __restrict__ alh, const float* __restrict__ arh,
                                      const float* __restrict__ h,
                                      const unsigned int* __restrict__ mu,
                                      const float* __restrict__ ssum,
                                      float* __restrict__ agg, int E)
{
    int e = blockIdx.x * (blockDim.x >> 5) + (threadIdx.x >> 5);
    if (e >= E) return;
    int lane = threadIdx.x & 31;
    int s = src[e], d = dst[e];
    const float4 a  = *(const float4*)(alh + (size_t)s * 128 + lane * 4);
    const float4 r  = *(const float4*)(arh + (size_t)d * 128 + lane * 4);
    const float4 hv = *(const float4*)(h   + (size_t)s * 128 + lane * 4);
    const uint4  m  = *(const uint4*)(mu   + (size_t)d * 128 + lane * 4);
    const float4 sv = *(const float4*)(ssum + (size_t)d * 128 + lane * 4);
    float* op = agg + (size_t)d * 128 + lane * 4;
    atomicAdd(op + 0, (__expf(lrelu(a.x + r.x) - decf(m.x)) / (sv.x + 1e-16f)) * hv.x);
    atomicAdd(op + 1, (__expf(lrelu(a.y + r.y) - decf(m.y)) / (sv.y + 1e-16f)) * hv.y);
    atomicAdd(op + 2, (__expf(lrelu(a.z + r.z) - decf(m.z)) / (sv.z + 1e-16f)) * hv.z);
    atomicAdd(op + 3, (__expf(lrelu(a.w + r.w) - decf(m.w)) / (sv.w + 1e-16f)) * hv.w);
}

// ---------------------------------------------------------------------------
// log_softmax over 40 classes; one wave per node, shuffle reductions.
// ---------------------------------------------------------------------------
__global__ void logsoftmax_kernel(const float* __restrict__ logits,
                                  float* __restrict__ out, int n, int C)
{
    int node = blockIdx.x * (blockDim.x >> 5) + (threadIdx.x >> 5);
    if (node >= n) return;
    int lane = threadIdx.x & 31;
    const float NEG_INF = __int_as_float(0xff800000);
    const float* p = logits + (size_t)node * C;
    float v0 = (lane      < C) ? p[lane]      : NEG_INF;
    float v1 = (lane + 32 < C) ? p[lane + 32] : NEG_INF;
    float m = fmaxf(v0, v1);
    for (int o = 16; o > 0; o >>= 1) m = fmaxf(m, __shfl_xor(m, o, 32));
    float s = ((lane < C) ? __expf(v0 - m) : 0.f) +
              ((lane + 32 < C) ? __expf(v1 - m) : 0.f);
    for (int o = 16; o > 0; o >>= 1) s += __shfl_xor(s, o, 32);
    float lse = __logf(s) + m;
    if (lane < C)      out[(size_t)node * C + lane]      = v0 - lse;
    if (lane + 32 < C) out[(size_t)node * C + lane + 32] = v1 - lse;
}

// ---------------------------------------------------------------------------
// Launch
// ---------------------------------------------------------------------------
extern "C" void kernel_launch(void* const* d_in, const int* in_sizes, int n_in,
                              void* d_out, int out_size, void* d_ws, size_t ws_size,
                              hipStream_t stream)
{
    const float* x   = (const float*)d_in[0];
    const int*   ei  = (const int*)d_in[1];
    const float* W[3]  = {(const float*)d_in[2],  (const float*)d_in[6],  (const float*)d_in[10]};
    const float* B[3]  = {(const float*)d_in[3],  (const float*)d_in[7],  (const float*)d_in[11]};
    const float* AL[3] = {(const float*)d_in[4],  (const float*)d_in[8],  (const float*)d_in[12]};
    const float* AR[3] = {(const float*)d_in[5],  (const float*)d_in[9],  (const float*)d_in[13]};
    const float* wp1 = (const float*)d_in[14];
    const float* bp1 = (const float*)d_in[15];
    const float* wp2 = (const float*)d_in[16];
    const float* bp2 = (const float*)d_in[17];

    const int N  = in_sizes[0] / 128;
    const int E  = in_sizes[1] / 2;
    const int HC = 128, CMID = 64, NOUT = 40;

    float* ws = (float*)d_ws;
    const size_t nhc = (size_t)N * HC;
    float*        h    = ws;
    float*        alh  = h + nhc;
    float*        arh  = alh + nhc;
    unsigned int* mu   = (unsigned int*)(arh + nhc);
    float*        ssum = (float*)(mu + nhc);
    float*        agg  = ssum + nhc;
    float*        p1   = alh;   // reuse after edge phases are done
    float*        p2   = arh;   // reuse
    float*        outp = (float*)d_out;

    const int* srcp = ei;
    const int* dstp = ei + E;

    dim3 gGemm((N + 15) / 16);
    const int EW = 8;                               // waves (edges) per block
    dim3 gEdge((E + EW - 1) / EW), bEdge(32 * EW);
    unsigned long long nelem = (unsigned long long)nhc;
    dim3 gInit((unsigned)((nelem + 255) / 256));

    const float* cur = x;
    for (int l = 0; l < 3; ++l) {
        gemm_wmma_kernel<<<gGemm, 256, 0, stream>>>(
            cur, W[l], B[l], AL[l], AR[l], h, alh, arh,
            N, 128, HC, HC, HC, (l > 0) ? 1 : 0);
        init_kernel<<<gInit, 256, 0, stream>>>(mu, ssum, agg, nelem);
        edge_max_kernel<<<gEdge, bEdge, 0, stream>>>(srcp, dstp, alh, arh, mu, E);
        edge_expsum_kernel<<<gEdge, bEdge, 0, stream>>>(srcp, dstp, alh, arh, mu, ssum, E);
        edge_aggregate_kernel<<<gEdge, bEdge, 0, stream>>>(srcp, dstp, alh, arh, h, mu, ssum, agg, E);
        cur = agg;
    }

    // post-MLP: relu(agg) @ wp1.T + bp1  -> [N,64]
    gemm_wmma_kernel<<<gGemm, 128, 0, stream>>>(
        agg, wp1, bp1, nullptr, nullptr, p1, nullptr, nullptr,
        N, 128, CMID, CMID, CMID, 1);
    // p1 @ wp2.T + bp2 -> [N,40] (NCpad = 48 -> 3 waves)
    gemm_wmma_kernel<<<gGemm, 96, 0, stream>>>(
        p1, wp2, bp2, nullptr, nullptr, p2, nullptr, nullptr,
        N, CMID, NOUT, 48, NOUT, 0);

    const int LW = 8;
    dim3 gLsm((N + LW - 1) / LW);
    logsoftmax_kernel<<<gLsm, 32 * LW, 0, stream>>>(p2, outp, N, NOUT);
}
